// MuchPool_14620068675980
// MI455X (gfx1250) — compile-verified, matrix operand order checked by reading, not run on hardware
//
#include <hip/hip_runtime.h>
#include <hip/hip_bf16.h>

// MuchPool for MI455X (gfx1250, wave32, WMMA).
//
// new_adj = S @ adj @ S^T with S = masked selected rows of the symmetric 0/1
// adjacency. All operands are 0/1 or small integer path counts -> exact in
// f16 inputs with f32 WMMA accumulation (v_wmma_f32_16x16x32_f16).
//
// Strategy (v2): pre-pack adj into a zero-padded fp16 buffer Ah[1024][1024]
// per batch (row 1023 and all rows/cols >= 1000 are zero). Then both GEMMs
// run with UNCONDITIONAL aligned b128 fragment loads: masked/invalid rows are
// realized by pointing the lane at the all-zero row 1023 (pointer select, no
// exec-mask branching, no in-loop cvt). adj symmetry makes every B-operand
// column a contiguous row load. Each wave computes a 16x64 strip (4
// accumulators) to reuse its A fragment across 4 WMMAs.

#define BATCH 16
#define NN    1000
#define DD    256
#define KMAX  512
#define KPAD  1024

typedef _Float16 v16h __attribute__((ext_vector_type(16)));
typedef _Float16 v8h  __attribute__((ext_vector_type(8)));
typedef _Float16 v4h  __attribute__((ext_vector_type(4)));
typedef float    v8f  __attribute__((ext_vector_type(8)));

// ---------------- workspace layout (bytes) ----------------
#define OFF_ORDER   ((size_t)0)          // 16*512*4   = 32768
#define OFF_NUNION  ((size_t)32768)      // 16*4
#define OFF_INV1    ((size_t)33024)      // 16*1000*4  = 64000
#define OFF_INV2    ((size_t)97280)      // 64000
#define OFF_M1      ((size_t)161280)     // 16*1000 (u8)
#define OFF_M2      ((size_t)177280)     // 16*1000 (u8)
#define OFF_AH      ((size_t)262144)     // 16*1024*1024*2 = 32 MiB (padded f16 adj)
#define OFF_TH      ((size_t)(262144 + (size_t)BATCH*KPAD*KPAD*2)) // 16 MiB (f16 T)

// ---------------------------------------------------------------------------
// Kernel 1: masks, inverse permutations, stable order, new_mask output.
// ---------------------------------------------------------------------------
__global__ __launch_bounds__(256)
void mp_setup(const int* __restrict__ idx1, const int* __restrict__ idx2,
              const int* __restrict__ k1,   const int* __restrict__ k2,
              int* __restrict__ order, int* __restrict__ nUnion,
              int* __restrict__ inv1,  int* __restrict__ inv2,
              unsigned char* __restrict__ m1, unsigned char* __restrict__ m2,
              float* __restrict__ outMask)
{
  const int b   = blockIdx.x;
  const int tid = threadIdx.x;
  const int kk1 = k1[b], kk2 = k2[b];

  for (int j = tid; j < NN; j += blockDim.x) {
    const int v1 = idx1[b * NN + j];
    m1[b * NN + v1]   = (j < kk1) ? 1 : 0;
    inv1[b * NN + v1] = j;
    const int v2 = idx2[b * NN + j];
    m2[b * NN + v2]   = (j < kk2) ? 1 : 0;
    inv2[b * NN + v2] = j;
  }
  __syncthreads();

  __shared__ int sNU;
  if (tid == 0) {
    // stable: union nodes ascending, then non-union ascending; take KMAX
    int cnt = 0;
    for (int v = 0; v < NN; ++v)
      if (m1[b * NN + v] | m2[b * NN + v]) {
        if (cnt < KMAX) order[b * KMAX + cnt] = v;
        ++cnt;
      }
    const int nu = cnt < KMAX ? cnt : KMAX;
    for (int v = 0; v < NN && cnt < KMAX; ++v)
      if (!(m1[b * NN + v] | m2[b * NN + v])) order[b * KMAX + cnt++] = v;
    nUnion[b] = nu;
    sNU = nu;
  }
  __syncthreads();

  const int nu = sNU;
  for (int i = tid; i < KMAX; i += blockDim.x)
    outMask[b * KMAX + i] = (i < nu) ? 1.0f : 0.0f;
}

// ---------------------------------------------------------------------------
// Kernel 2: pack adj (f32, 1000x1000) -> Ah (f16, 1024x1024, zero padded).
// One block per (b,row); thread t covers columns 4t..4t+3.
// ---------------------------------------------------------------------------
__global__ __launch_bounds__(256)
void mp_pack(const float* __restrict__ adj, _Float16* __restrict__ Ah)
{
  const int row = blockIdx.x & (KPAD - 1);
  const int b   = blockIdx.x >> 10;
  const int c0  = threadIdx.x * 4;            // 0..1020, multiple of 4

  v4h o;
  if (row < NN && c0 < NN) {                  // c0 <= 996 -> covers up to col 999
    const float4 x = *(const float4*)(adj + ((size_t)b * NN + row) * NN + c0);
    o[0] = (_Float16)x.x; o[1] = (_Float16)x.y;
    o[2] = (_Float16)x.z; o[3] = (_Float16)x.w;
  } else {
    o[0] = o[1] = o[2] = o[3] = (_Float16)0.0f;
  }
  *(v4h*)(Ah + ((size_t)b * KPAD + row) * KPAD + c0) = o;
}

// ---------------------------------------------------------------------------
// Kernel 3: new_H gather + merge.  One block per (b,i) slot; thread per d.
// ---------------------------------------------------------------------------
__global__ __launch_bounds__(256)
void mp_newh(const float* __restrict__ H1, const float* __restrict__ H2,
             const int* __restrict__ order, const int* __restrict__ nUnion,
             const int* __restrict__ inv1,  const int* __restrict__ inv2,
             const unsigned char* __restrict__ m1, const unsigned char* __restrict__ m2,
             float* __restrict__ outH)
{
  const int i = blockIdx.x & (KMAX - 1);
  const int b = blockIdx.x >> 9;
  const int d = threadIdx.x;
  const size_t o = ((size_t)b * KMAX + i) * DD + d;

  if (i >= nUnion[b]) { outH[o] = 0.0f; return; }

  const int v = order[b * KMAX + i];
  const bool in1 = m1[b * NN + v] != 0;
  const bool in2 = m2[b * NN + v] != 0;
  const float h1 = in1 ? H1[((size_t)b * NN + inv1[b * NN + v]) * DD + d] : 0.0f;
  const float h2 = in2 ? H2[((size_t)b * NN + inv2[b * NN + v]) * DD + d] : 0.0f;
  outH[o] = (in1 && in2) ? 0.5f * (h1 + h2) : (h1 + h2);
}

// ---------------------------------------------------------------------------
// f16 fragment load (ISA 7.12.2 16-bit A/B layout, wave32):
//   lane L: row/col index = base + (L & 15); hi = L >> 4
//   halves 0..7  = K = kB + hi*8      + {0..7}
//   halves 8..15 = K = kB + 16 + hi*8 + {0..7}
// Rows are KPAD-sized and fully valid -> two unconditional 16B loads.
// ---------------------------------------------------------------------------
__device__ __forceinline__
v16h frag16(const _Float16* __restrict__ row, int kB, int hi)
{
  const v8h lo = *(const v8h*)(row + kB + hi * 8);
  const v8h up = *(const v8h*)(row + kB + 16 + hi * 8);
  v16h a;
  #pragma unroll
  for (int q = 0; q < 8; ++q) { a[q] = lo[q]; a[q + 8] = up[q]; }
  return a;
}

// ---------------------------------------------------------------------------
// Kernel 4 (GEMM1): T = S @ adj  (M=512, N=1024, K=1024), f16 out.
// Wave computes 16x64 (4 accumulators, A fragment reused 4x). S rows are
// gathered via order[]; masked-out lanes read the all-zero row 1023.
// ---------------------------------------------------------------------------
__global__ __launch_bounds__(256)
void mp_gemm1(const _Float16* __restrict__ Ah, const int* __restrict__ order,
              const int* __restrict__ nUnion, _Float16* __restrict__ Th)
{
  const int b      = blockIdx.z;
  const int mTile  = blockIdx.y;                    // 0..31
  const int wave   = threadIdx.x >> 5;              // 0..7
  const int nGroup = blockIdx.x * 8 + wave;         // 0..15 (x 64 cols)
  const int lane   = threadIdx.x & 31;
  const int r  = lane & 15;
  const int hi = lane >> 4;

  const _Float16* Ab      = Ah + (size_t)b * KPAD * KPAD;
  const _Float16* zeroRow = Ab + (size_t)(KPAD - 1) * KPAD;   // all zeros

  const int i  = mTile * 16 + r;
  const int nu = nUnion[b];
  const int oi = order[b * KMAX + i];
  const _Float16* rowA = (i < nu) ? (Ab + (size_t)oi * KPAD) : zeroRow;

  const _Float16* rowB[4];
  #pragma unroll
  for (int t = 0; t < 4; ++t)
    rowB[t] = Ab + (size_t)((nGroup * 4 + t) * 16 + r) * KPAD; // rows>=1000 are zero

  v8f c[4];
  #pragma unroll
  for (int t = 0; t < 4; ++t)
    #pragma unroll
    for (int q = 0; q < 8; ++q) c[t][q] = 0.0f;

  for (int kB = 0; kB < KPAD; kB += 32) {
    __builtin_prefetch(rowA + kB + 256, 0, 1);      // global_prefetch_b8
    const v16h a = frag16(rowA, kB, hi);
    #pragma unroll
    for (int t = 0; t < 4; ++t) {
      const v16h bb = frag16(rowB[t], kB, hi);
      c[t] = __builtin_amdgcn_wmma_f32_16x16x32_f16(false, a, false, bb,
                                                    (short)0, c[t], false, false);
    }
  }

  // C/D layout: VGPR v -> M = v + 8*hi, N = lane & 15
  _Float16* Tb = Th + (size_t)b * KMAX * KPAD;
  #pragma unroll
  for (int t = 0; t < 4; ++t) {
    const int col = (nGroup * 4 + t) * 16 + r;
    #pragma unroll
    for (int v = 0; v < 8; ++v) {
      const int row = mTile * 16 + v + 8 * hi;
      Tb[(size_t)row * KPAD + col] = (_Float16)c[t][v];
    }
  }
}

// ---------------------------------------------------------------------------
// Kernel 5 (GEMM2): new_adj = T @ S^T  (M=512, N=512, K=1024), f32 out.
// ---------------------------------------------------------------------------
__global__ __launch_bounds__(256)
void mp_gemm2(const _Float16* __restrict__ Ah, const int* __restrict__ order,
              const int* __restrict__ nUnion, const _Float16* __restrict__ Th,
              float* __restrict__ outAdj)
{
  const int b      = blockIdx.z;
  const int mTile  = blockIdx.y;                    // 0..31
  const int nGroup = threadIdx.x >> 5;              // 0..7 (x 64 cols)
  const int lane   = threadIdx.x & 31;
  const int r  = lane & 15;
  const int hi = lane >> 4;

  const _Float16* Ab      = Ah + (size_t)b * KPAD * KPAD;
  const _Float16* zeroRow = Ab + (size_t)(KPAD - 1) * KPAD;

  const _Float16* rowA = Th + ((size_t)b * KMAX + mTile * 16 + r) * KPAD;

  const int nu = nUnion[b];
  const _Float16* rowB[4];
  #pragma unroll
  for (int t = 0; t < 4; ++t) {
    const int j  = (nGroup * 4 + t) * 16 + r;       // 0..511
    const int oj = order[b * KMAX + j];
    rowB[t] = (j < nu) ? (Ab + (size_t)oj * KPAD) : zeroRow;
  }

  v8f c[4];
  #pragma unroll
  for (int t = 0; t < 4; ++t)
    #pragma unroll
    for (int q = 0; q < 8; ++q) c[t][q] = 0.0f;

  for (int kB = 0; kB < KPAD; kB += 32) {
    __builtin_prefetch(rowA + kB + 256, 0, 1);
    const v16h a = frag16(rowA, kB, hi);
    #pragma unroll
    for (int t = 0; t < 4; ++t) {
      const v16h bb = frag16(rowB[t], kB, hi);
      c[t] = __builtin_amdgcn_wmma_f32_16x16x32_f16(false, a, false, bb,
                                                    (short)0, c[t], false, false);
    }
  }

  float* Ob = outAdj + (size_t)b * KMAX * KMAX;
  #pragma unroll
  for (int t = 0; t < 4; ++t) {
    const int col = (nGroup * 4 + t) * 16 + r;
    #pragma unroll
    for (int v = 0; v < 8; ++v) {
      const int row = mTile * 16 + v + 8 * hi;
      Ob[(size_t)row * KMAX + col] = c[t][v];
    }
  }
}

// ---------------------------------------------------------------------------
extern "C" void kernel_launch(void* const* d_in, const int* in_sizes, int n_in,
                              void* d_out, int out_size, void* d_ws, size_t ws_size,
                              hipStream_t stream)
{
  const float* adj  = (const float*)d_in[0];
  const float* H1   = (const float*)d_in[1];
  const float* H2   = (const float*)d_in[2];
  const int*   idx1 = (const int*)d_in[3];
  const int*   idx2 = (const int*)d_in[4];
  const int*   k1   = (const int*)d_in[5];
  const int*   k2   = (const int*)d_in[6];

  float* out     = (float*)d_out;
  float* outH    = out;                                        // [16,512,256]
  float* outAdj  = out + (size_t)BATCH * KMAX * DD;            // [16,512,512]
  float* outMask = outAdj + (size_t)BATCH * KMAX * KMAX;       // [16,512]

  char* ws = (char*)d_ws;
  int*           order  = (int*)(ws + OFF_ORDER);
  int*           nUnion = (int*)(ws + OFF_NUNION);
  int*           inv1   = (int*)(ws + OFF_INV1);
  int*           inv2   = (int*)(ws + OFF_INV2);
  unsigned char* m1     = (unsigned char*)(ws + OFF_M1);
  unsigned char* m2     = (unsigned char*)(ws + OFF_M2);
  _Float16*      Ah     = (_Float16*)(ws + OFF_AH);
  _Float16*      Th     = (_Float16*)(ws + OFF_TH);

  mp_setup<<<BATCH, 256, 0, stream>>>(idx1, idx2, k1, k2, order, nUnion,
                                      inv1, inv2, m1, m2, outMask);

  mp_pack<<<BATCH * KPAD, 256, 0, stream>>>(adj, Ah);

  mp_newh<<<BATCH * KMAX, 256, 0, stream>>>(H1, H2, order, nUnion,
                                            inv1, inv2, m1, m2, outH);

  dim3 g1(2, 32, BATCH);   // 16 nGroups (8 waves/block) x 32 mTiles x B
  mp_gemm1<<<g1, 256, 0, stream>>>(Ah, order, nUnion, Th);

  dim3 g2(1, 32, BATCH);   // 8 nGroups (8 waves/block) x 32 mTiles x B
  mp_gemm2<<<g2, 256, 0, stream>>>(Ah, order, nUnion, Th, outAdj);
}
